// SVDNet_10960756540133
// MI455X (gfx1250) — matrix operand order repaired and checked by reading, not compiled, not run on hardware
//
#include <hip/hip_runtime.h>
#include <math.h>

typedef _Float16 f16;
typedef __attribute__((ext_vector_type(16))) _Float16 v16h;
typedef __attribute__((ext_vector_type(8)))  _Float16 v8h;
typedef __attribute__((ext_vector_type(8)))  float    v8f;

#define PW  520         // padded width (interior cols 1..512)
#define PHA 518         // padded rows (interior rows 1..512; extra tail rows for conv1 K-pad over-read)

// ---------------- utility kernels ----------------

__global__ void zero_u32(unsigned int* p, long long n) {
    long long i = (long long)blockIdx.x * blockDim.x + threadIdx.x;
    if (i < n) p[i] = 0u;
}

// x (512,512,2) f32 -> channel-last padded f16 [PHA][PW][2]
__global__ void pack_x(const float* __restrict__ x, f16* __restrict__ xpad) {
    int i = blockIdx.x * blockDim.x + threadIdx.x;   // 512*512*2
    if (i >= 512 * 512 * 2) return;
    int ch  = i & 1;
    int col = (i >> 1) & 511;
    int row = i >> 10;
    xpad[((size_t)(row + 1) * PW + (col + 1)) * 2 + ch] = (f16)x[i];
}

// weights (Cout, Cin, 3, 3) f32 -> f16 [Cout][Kpad] with K reordered as (cell=ky*3+kx, cin)
__global__ void pack_w(const float* __restrict__ w, f16* __restrict__ wf,
                       int Cout, int Cin, int Kpad) {
    int i = blockIdx.x * blockDim.x + threadIdx.x;
    if (i >= Cout * Kpad) return;
    int K = Cin * 9;
    int co = i / Kpad, kp = i % Kpad;
    f16 v = (f16)0.f;
    if (kp < K) {
        int cell = kp / Cin;          // ky*3+kx
        int cin  = kp % Cin;
        v = (f16)w[co * K + cin * 9 + cell];
    }
    wf[i] = v;
}

// ---------------- WMMA implicit-GEMM conv + ReLU (channel-last) ----------------
// wave tile: 64 pixels (4 x 16 along W) x 16 output channels; K chunks of 32.
template<int CIN, int COUT, int KPAD>
__global__ __launch_bounds__(256)
void conv_relu_wmma(const f16* __restrict__ in, const f16* __restrict__ wf,
                    const float* __restrict__ bias, f16* __restrict__ out) {
    constexpr int K   = CIN * 9;
    constexpr int NCT = COUT / 16;
    const int wave = threadIdx.x >> 5;
    const int lane = threadIdx.x & 31;
    const int tile = blockIdx.x * 8 + wave;
    const int ct   = tile % NCT;
    const int p    = tile / NCT;
    const int y    = p >> 3;              // output row 0..511
    const int x0   = (p & 7) << 6;        // output col base (64-pixel tile)
    const int m    = lane & 15;
    const int half = lane >> 4;

    v8f acc[4] = {};
    const f16* wrow = wf + (size_t)(ct * 16 + m) * KPAD + half * 8;

    for (int kk = 0; kk < K; kk += 32) {
        // B fragment: two 8-half runs (lane = N; K kk+{0..7}+half*8 and kk+16+{0..7}+half*8)
        v16h b = __builtin_shufflevector(*(const v8h*)(wrow + kk), *(const v8h*)(wrow + kk + 16),
                  0,1,2,3,4,5,6,7,8,9,10,11,12,13,14,15);
        __builtin_prefetch(wrow + kk + 32, 0, 0);

        if constexpr (CIN >= 32) {
            // chunk lies inside one (ky,kx) cell; contiguous along cin
            const int cell = kk / CIN;
            const int cin0 = kk % CIN;
            const int ky = cell / 3, kx = cell - ky * 3;
            const f16* base = in + ((size_t)(y + ky) * PW + (x0 + kx)) * CIN
                                 + cin0 + half * 8;
#pragma unroll
            for (int s = 0; s < 4; ++s) {
                const f16* pa = base + (size_t)(s * 16 + m) * CIN;
                v8h lo = *(const v8h*)pa;          // K kk+{0..7}+half*8
                v8h hi = *(const v8h*)(pa + 16);   // K kk+16+{0..7}+half*8
                v16h a = __builtin_shufflevector(lo, hi,
                          0,1,2,3,4,5,6,7,8,9,10,11,12,13,14,15);
                acc[s] = __builtin_amdgcn_wmma_f32_16x16x32_f16(
                             false, a, false, b, (short)0, acc[s], false, false);
            }
        } else {
            // small-Cin gather path (conv1). No K guard: weight columns k>=K are
            // zero-padded, and over-read activations are finite (buffer zero-filled,
            // PHA=518 covers cell<=15 -> ky<=5), so a * 0 = 0.
#pragma unroll
            for (int s = 0; s < 4; ++s) {
                v16h a;
#pragma unroll
                for (int t = 0; t < 16; ++t) {
                    int k = kk + (t < 8 ? t + half * 8 : t + 8 + half * 8);
                    int cell = k / CIN;            // CIN=2 -> shift
                    int cin  = k % CIN;
                    int ky = cell / 3, kx = cell - ky * 3;
                    a[t] = in[((size_t)(y + ky) * PW + (x0 + s * 16 + m + kx)) * CIN + cin];
                }
                acc[s] = __builtin_amdgcn_wmma_f32_16x16x32_f16(
                             false, a, false, b, (short)0, acc[s], false, false);
            }
        }
    }

    // epilogue: lane holds channel ct*16+m for 8 pixels per subtile
    const int co = ct * 16 + m;
    const float bv = bias[co];
#pragma unroll
    for (int s = 0; s < 4; ++s) {
        size_t base = ((size_t)(y + 1) * PW + (x0 + s * 16 + 1)) * COUT + co;
#pragma unroll
        for (int r = 0; r < 8; ++r) {
            float o = acc[s][r] + bv;
            o = fmaxf(o, 0.f);
            out[base + (size_t)(r + half * 8) * COUT] = (f16)o;
        }
    }
}

// ---------------- global average pool (channel-last, deterministic 2-stage) ----------------
__global__ void avgpool_partial(const f16* __restrict__ c3, float* __restrict__ part) {
    int ch  = threadIdx.x & 127;     // lanes -> consecutive channels: coalesced
    int sub = threadIdx.x >> 7;      // 0/1
    float s = 0.f;
    int p0 = blockIdx.x * 1024;      // 256 blocks x 1024 pixels
    for (int i = sub; i < 1024; i += 2) {
        int p = p0 + i;
        int yy = p >> 9, xx = p & 511;
        s += (float)c3[((size_t)(yy + 1) * PW + (xx + 1)) * 128 + ch];
    }
    __shared__ float red[128];
    if (sub) red[ch] = s;
    __syncthreads();
    if (!sub) part[blockIdx.x * 128 + ch] = s + red[ch];
}

__global__ void avgpool_final(const float* __restrict__ part, float* __restrict__ g) {
    int ch = threadIdx.x;            // 128 threads
    float s = 0.f;
    for (int b = 0; b < 256; ++b) s += part[b * 128 + ch];
    g[ch] = s * (1.f / (512.f * 512.f));
}

// ---------------- GEMV y = act(W x + b), one wave per row, float4 loads ----------------
// all cols are multiples of 128 -> each lane handles cols/128 float4 chunks
__global__ void gemv_act(const float* __restrict__ W, const float* __restrict__ x,
                         const float* __restrict__ b, float* __restrict__ y,
                         int rows, int cols, int act) {
    int row = blockIdx.x * (blockDim.x >> 5) + (threadIdx.x >> 5);
    if (row >= rows) return;
    int lane = threadIdx.x & 31;
    const float4* w4 = (const float4*)(W + (size_t)row * cols);
    const float4* x4 = (const float4*)x;
    int c4 = cols >> 2;
    float s = 0.f;
    for (int c = lane; c < c4; c += 32) {
        float4 wv = w4[c];
        float4 xv = x4[c];
        s += wv.x * xv.x + wv.y * xv.y + wv.z * xv.z + wv.w * xv.w;
    }
    for (int off = 16; off; off >>= 1) s += __shfl_down(s, off, 32);
    if (lane == 0) {
        s += b[row];
        if (act == 1) s = fmaxf(s, 0.f);
        else if (act == 2) s = (s > 20.f) ? s : log1pf(expf(s));
        y[row] = s;
    }
}

// ---------------- scale + bitonic sort (descending) for s_values ----------------
__global__ void sort_s(const float* __restrict__ slog, const float* __restrict__ s_temp,
                       float* __restrict__ out) {
    __shared__ float a[64];
    int t = threadIdx.x;      // 64 threads
    a[t] = slog[t] * fabsf(s_temp[0]);
    __syncthreads();
    for (int k = 2; k <= 64; k <<= 1) {
        for (int j = k >> 1; j > 0; j >>= 1) {
            int ixj = t ^ j;
            if (ixj > t) {
                bool up = ((t & k) == 0);
                float xa = a[t], xb = a[ixj];
                if (up ? (xa > xb) : (xa < xb)) { a[t] = xb; a[ixj] = xa; }
            }
            __syncthreads();
        }
    }
    out[t] = a[63 - t];       // ascending -> reversed = descending
}

// ---------------- complex orthogonalization ----------------
// 512-thread block; 16 waves. Two-level shuffle reduction, LDS broadcast.
__device__ __forceinline__ void block_reduce2(float& a, float& b, float* sb) {
    for (int off = 16; off; off >>= 1) {
        a += __shfl_down(a, off, 32);
        b += __shfl_down(b, off, 32);
    }
    int w = threadIdx.x >> 5;
    if ((threadIdx.x & 31) == 0) { sb[w * 2] = a; sb[w * 2 + 1] = b; }
    __syncthreads();
    if (threadIdx.x < 32) {
        int l = threadIdx.x;
        float ta = (l < 16) ? sb[l * 2]     : 0.f;
        float tb = (l < 16) ? sb[l * 2 + 1] : 0.f;
        for (int off = 8; off; off >>= 1) {
            ta += __shfl_down(ta, off, 32);
            tb += __shfl_down(tb, off, 32);
        }
        if (l == 0) { sb[0] = ta; sb[1] = tb; }
    }
    __syncthreads();
    a = sb[0]; b = sb[1];
    __syncthreads();
}

// sequential modified Gram-Schmidt; blockIdx.x: 0 = u-matrix, 1 = v-matrix
__global__ __launch_bounds__(512)
void mgs_kernel(const float* __restrict__ uraw, const float* __restrict__ vraw,
                float* __restrict__ Qu, float* __restrict__ Qv) {
    const float* X = blockIdx.x ? vraw : uraw;
    float* Q = blockIdx.x ? Qv : Qu;
    __shared__ float sb[32];
    int t = threadIdx.x;   // row 0..511
    for (int j = 0; j < 64; ++j) {
        float vr = X[(t * 64 + j) * 2], vi = X[(t * 64 + j) * 2 + 1];
        for (int i = 0; i < j; ++i) {
            float qr = Q[(t * 64 + i) * 2], qi = Q[(t * 64 + i) * 2 + 1];
            float cr = qr * vr + qi * vi;     // conj(q) * v
            float ci = qr * vi - qi * vr;
            block_reduce2(cr, ci, sb);
            vr -= cr * qr - ci * qi;          // v -= c * q
            vi -= cr * qi + ci * qr;
        }
        float nn = vr * vr + vi * vi, d = 0.f;
        block_reduce2(nn, d, sb);
        float inv = 1.f / sqrtf(nn + 1e-10f);
        Q[(t * 64 + j) * 2]     = vr * inv;
        Q[(t * 64 + j) * 2 + 1] = vi * inv;
    }
}

// one refinement sweep: column-parallel, reads old Q. grid = (64 cols, 2 matrices)
__global__ __launch_bounds__(512)
void refine_kernel(const float* __restrict__ Qu_in, const float* __restrict__ Qv_in,
                   float* __restrict__ Qu_out, float* __restrict__ Qv_out) {
    const float* Qin = blockIdx.y ? Qv_in : Qu_in;
    float* Qout = blockIdx.y ? Qv_out : Qu_out;
    int j = blockIdx.x;
    __shared__ float sb[32];
    int t = threadIdx.x;
    float vr = Qin[(t * 64 + j) * 2], vi = Qin[(t * 64 + j) * 2 + 1];
    for (int i = 0; i < 64; ++i) {
        float qr = Qin[(t * 64 + i) * 2], qi = Qin[(t * 64 + i) * 2 + 1];
        float cr = qr * vr + qi * vi;
        float ci = qr * vi - qi * vr;
        block_reduce2(cr, ci, sb);
        float sc = (i == j) ? 0.f : 0.5f;
        vr -= sc * (cr * qr - ci * qi);
        vi -= sc * (cr * qi + ci * qr);
    }
    float nn = vr * vr + vi * vi, d = 0.f;
    block_reduce2(nn, d, sb);
    float inv = 1.f / sqrtf(nn + 1e-10f);
    Qout[(t * 64 + j) * 2]     = vr * inv;
    Qout[(t * 64 + j) * 2 + 1] = vi * inv;
}

// ---------------- host launch ----------------

extern "C" void kernel_launch(void* const* d_in, const int* in_sizes, int n_in,
                              void* d_out, int out_size, void* d_ws, size_t ws_size,
                              hipStream_t stream) {
    const float* x       = (const float*)d_in[0];
    const float* conv1_w = (const float*)d_in[1];  const float* conv1_b = (const float*)d_in[2];
    const float* conv2_w = (const float*)d_in[3];  const float* conv2_b = (const float*)d_in[4];
    const float* conv3_w = (const float*)d_in[5];  const float* conv3_b = (const float*)d_in[6];
    const float* fc1_w   = (const float*)d_in[7];  const float* fc1_b   = (const float*)d_in[8];
    const float* fc2_w   = (const float*)d_in[9];  const float* fc2_b   = (const float*)d_in[10];
    const float* s1_w    = (const float*)d_in[11]; const float* s1_b    = (const float*)d_in[12];
    const float* s2_w    = (const float*)d_in[13]; const float* s2_b    = (const float*)d_in[14];
    const float* u1_w    = (const float*)d_in[15]; const float* u1_b    = (const float*)d_in[16];
    const float* u2_w    = (const float*)d_in[17]; const float* u2_b    = (const float*)d_in[18];
    const float* v1_w    = (const float*)d_in[19]; const float* v1_b    = (const float*)d_in[20];
    const float* v2_w    = (const float*)d_in[21]; const float* v2_b    = (const float*)d_in[22];
    const float* s_temp  = (const float*)d_in[23];
    float* out = (float*)d_out;

    char* ws = (char*)d_ws;
    size_t off = 0;
    auto alloc = [&](size_t bytes) -> char* {
        off = (off + 255) & ~(size_t)255;
        char* p = ws + off;
        off += bytes;
        return p;
    };

    const size_t CELLS = (size_t)PHA * PW;   // padded pixels per buffer
    f16* xpad = (f16*)alloc(CELLS * 2   * sizeof(f16));
    f16* c1b  = (f16*)alloc(CELLS * 32  * sizeof(f16));
    f16* c2b  = (f16*)alloc(CELLS * 64  * sizeof(f16));
    f16* c3b  = (f16*)alloc(CELLS * 128 * sizeof(f16));
    char* planes_end = ws + off;

    f16* wf1 = (f16*)alloc(32  * 32  * sizeof(f16));
    f16* wf2 = (f16*)alloc(64  * 288 * sizeof(f16));
    f16* wf3 = (f16*)alloc(128 * 576 * sizeof(f16));

    float* part = (float*)alloc(256 * 128 * 4);
    float* g    = (float*)alloc(128  * 4);
    float* h1   = (float*)alloc(256  * 4);
    float* feat = (float*)alloc(512  * 4);
    float* sh   = (float*)alloc(256  * 4);
    float* slog = (float*)alloc(64   * 4);
    float* uh   = (float*)alloc(1024 * 4);
    float* vh   = (float*)alloc(1024 * 4);
    float* uraw = (float*)alloc(65536 * 4);
    float* vraw = (float*)alloc(65536 * 4);
    float* Qu0  = (float*)alloc(65536 * 4);
    float* Qv0  = (float*)alloc(65536 * 4);
    float* Qu1  = (float*)alloc(65536 * 4);
    float* Qv1  = (float*)alloc(65536 * 4);

    // zero all padded activation buffers (pads must be zero every call)
    long long nz = (long long)(planes_end - (char*)xpad) / 4;
    zero_u32<<<(unsigned)((nz + 255) / 256), 256, 0, stream>>>((unsigned int*)xpad, nz);

    pack_x<<<(512 * 512 * 2 + 255) / 256, 256, 0, stream>>>(x, xpad);
    pack_w<<<(32 * 32   + 255) / 256, 256, 0, stream>>>(conv1_w, wf1, 32,  2,  32);
    pack_w<<<(64 * 288  + 255) / 256, 256, 0, stream>>>(conv2_w, wf2, 64,  32, 288);
    pack_w<<<(128 * 576 + 255) / 256, 256, 0, stream>>>(conv3_w, wf3, 128, 64, 576);

    conv_relu_wmma<2,  32,  32 ><<<1024, 256, 0, stream>>>(xpad, wf1, conv1_b, c1b);
    conv_relu_wmma<32, 64,  288><<<2048, 256, 0, stream>>>(c1b,  wf2, conv2_b, c2b);
    conv_relu_wmma<64, 128, 576><<<4096, 256, 0, stream>>>(c2b,  wf3, conv3_b, c3b);

    avgpool_partial<<<256, 256, 0, stream>>>(c3b, part);
    avgpool_final<<<1, 128, 0, stream>>>(part, g);

    gemv_act<<<32,   256, 0, stream>>>(fc1_w, g,    fc1_b, h1,   256,   128,  1);
    gemv_act<<<64,   256, 0, stream>>>(fc2_w, h1,   fc2_b, feat, 512,   256,  1);
    gemv_act<<<32,   256, 0, stream>>>(s1_w,  feat, s1_b,  sh,   256,   512,  1);
    gemv_act<<<8,    256, 0, stream>>>(s2_w,  sh,   s2_b,  slog, 64,    256,  2);
    sort_s<<<1, 64, 0, stream>>>(slog, s_temp, out + 65536);

    gemv_act<<<128,  256, 0, stream>>>(u1_w, feat, u1_b, uh,   1024,  512,  1);
    gemv_act<<<8192, 256, 0, stream>>>(u2_w, uh,   u2_b, uraw, 65536, 1024, 0);
    gemv_act<<<128,  256, 0, stream>>>(v1_w, feat, v1_b, vh,   1024,  512,  1);
    gemv_act<<<8192, 256, 0, stream>>>(v2_w, vh,   v2_b, vraw, 65536, 1024, 0);

    mgs_kernel<<<2, 512, 0, stream>>>(uraw, vraw, Qu0, Qv0);
    dim3 rg(64, 2);
    refine_kernel<<<rg, 512, 0, stream>>>(Qu0, Qv0, Qu1, Qv1);
    refine_kernel<<<rg, 512, 0, stream>>>(Qu1, Qv1, Qu0, Qv0);
    refine_kernel<<<rg, 512, 0, stream>>>(Qu0, Qv0, out, out + 65600);
}
